// selfAttention_31585189495419
// MI455X (gfx1250) — matrix operand order repaired and checked by reading, not compiled, not run on hardware
//
#include <hip/hip_runtime.h>
#include <hip/hip_bf16.h>
#include <math.h>

typedef __attribute__((ext_vector_type(16))) __bf16 v16bf;
typedef __attribute__((ext_vector_type(8)))  float  v8f;
typedef __attribute__((ext_vector_type(16))) short  v16s;
typedef __attribute__((ext_vector_type(8)))  short  v8s;

#define DIM   128
#define NNB   50
#define MT    64      // neighbour rows padded to 4 M-tiles
#define PITCH 136     // shorts per LDS row: 16B-aligned rows + bank skew

__device__ __forceinline__ unsigned short f32_to_bf16_rne(float f) {
  unsigned u = __float_as_uint(f);
  unsigned r = u + 0x7FFFu + ((u >> 16) & 1u);
  return (unsigned short)(r >> 16);
}

// Offset (in shorts) of weight element (k = K-dim 0..127, d = N-dim 0..127)
// inside the pre-swizzled B-fragment array. Fragment = 512 shorts = one
// wave-wide 32x16 bf16 WMMA B operand: lane = (K>=16)*16 + (d&15), elem = K&15.
__device__ __forceinline__ int frag_off(int k, int d) {
  int nt = d >> 4, ks = k >> 5, kk = k & 31;
  int lane = ((kk >> 4) << 4) + (d & 15);
  return (nt * 4 + ks) * 512 + lane * 16 + (kk & 15);
}

// One-time prep: Wc = W @ A_w_top (f32), Wd = W @ A_w_bot (bf16 fragments),
// W itself as bf16 fragments. 16384 threads, 128-MAC loop each.
__global__ void gat_prep(const float* __restrict__ W, const float* __restrict__ Aw,
                         unsigned short* __restrict__ wF,
                         unsigned short* __restrict__ wdF,
                         float* __restrict__ Wc) {
  int t = blockIdx.x * blockDim.x + threadIdx.x;
  if (t >= DIM * DIM) return;
  int k = t >> 7, d = t & (DIM - 1);
  float accC = 0.f, accD = 0.f;
  for (int j = 0; j < DIM; ++j) {
    float w = W[k * DIM + j];
    accC = fmaf(w, Aw[j * DIM + d], accC);
    accD = fmaf(w, Aw[(DIM + j) * DIM + d], accD);
  }
  Wc[k * DIM + d] = accC;
  int off = frag_off(k, d);
  wF[off]  = f32_to_bf16_rne(W[k * DIM + d]);
  wdF[off] = f32_to_bf16_rne(accD);
}

// Fused GAT layer: 1 block = 1 batch row, 8 waves; wave w owns 16 output cols.
__global__ void __launch_bounds__(256) gat_main(
    const float* __restrict__ X, const float* __restrict__ Nb,
    const float* __restrict__ Ab, const unsigned short* __restrict__ wF,
    const unsigned short* __restrict__ wdF, const float* __restrict__ Wc,
    float* __restrict__ out) {
  __shared__ unsigned short aT[MT * PITCH];
  __shared__ float xsh[DIM];
  __shared__ float ush[DIM];

  const int b = blockIdx.x;
  const int t = threadIdx.x;

  if (t < DIM) xsh[t] = X[(long)b * DIM + t];

  // Stage neighbours to LDS as bf16, zero-pad rows >= NNB.
  for (int i = t; i < MT * DIM; i += 256) {
    int n = i >> 7, c = i & (DIM - 1);
    float v = (n < NNB) ? Nb[((long)b * NNB + n) * DIM + c] : 0.f;
    aT[n * PITCH + c] = f32_to_bf16_rne(v);
  }
  __syncthreads();

  // u[d] = x @ Wc + bias  (= px @ A_w_top + b), kept in f32 for accuracy.
  if (t < DIM) {
    float acc = Ab[t];
    for (int k = 0; k < DIM; ++k) acc = fmaf(xsh[k], Wc[k * DIM + t], acc);
    ush[t] = acc;
  }
  __syncthreads();

  const int w    = t >> 5;      // wave id: columns [16w, 16w+16)
  const int lane = t & 31;
  const int lrow = lane & 15;
  const int hi   = lane >> 4;   // A/C fragment half-lane group

  v8f accP[4] = {};   // P = Nb @ W
  v8f accE[4] = {};   // E = Nb @ Wd

  for (int ks = 0; ks < 4; ++ks) {
    v16bf bW  = __builtin_bit_cast(v16bf,
                  *(const v16s*)(wF  + (w * 4 + ks) * 512 + lane * 16));
    v16bf bWd = __builtin_bit_cast(v16bf,
                  *(const v16s*)(wdF + (w * 4 + ks) * 512 + lane * 16));
#pragma unroll
    for (int m = 0; m < 4; ++m) {
      // A fragment: lanes 0-15 rows m*16+lrow K{0..15}; lanes 16-31 K{16..31}
      const unsigned short* p = &aT[(m * 16 + lrow) * PITCH + ks * 32 + hi * 16];
      v8s lo8 = *(const v8s*)(p);
      v8s hi8 = *(const v8s*)(p + 8);
      v16s av = __builtin_shufflevector(lo8, hi8,
                  0, 1, 2, 3, 4, 5, 6, 7, 8, 9, 10, 11, 12, 13, 14, 15);
      v16bf a = __builtin_bit_cast(v16bf, av);
      accP[m] = __builtin_amdgcn_wmma_f32_16x16x32_bf16(
                  false, a, false, bW,  (short)0, accP[m], false, false);
      accE[m] = __builtin_amdgcn_wmma_f32_16x16x32_bf16(
                  false, a, false, bWd, (short)0, accE[m], false, false);
    }
  }

  // e = leaky_relu(E + u, 0.2); softmax over n (mask padded rows); out = sum a*P.
  // C-fragment: VGPR r holds row m*16 + hi*8 + r for column d = 16w + lrow.
  const float u = ush[w * 16 + lrow];
  float mx = -INFINITY;
#pragma unroll
  for (int m = 0; m < 4; ++m)
#pragma unroll
    for (int r = 0; r < 8; ++r) {
      int n = m * 16 + hi * 8 + r;
      float ev = accE[m][r] + u;
      ev = (ev > 0.f) ? ev : 0.2f * ev;
      accE[m][r] = ev;
      if (n < NNB) mx = fmaxf(mx, ev);
    }
  mx = fmaxf(mx, __shfl_xor(mx, 16, 32));

  float s = 0.f, o = 0.f;
#pragma unroll
  for (int m = 0; m < 4; ++m)
#pragma unroll
    for (int r = 0; r < 8; ++r) {
      int n = m * 16 + hi * 8 + r;
      if (n < NNB) {
        float a = __expf(accE[m][r] - mx);
        s += a;
        o += a * accP[m][r];
      }
    }
  s += __shfl_xor(s, 16, 32);
  o += __shfl_xor(o, 16, 32);

  if (hi == 0) out[(long)b * DIM + w * 16 + lrow] = o / s;
}

extern "C" void kernel_launch(void* const* d_in, const int* in_sizes, int n_in,
                              void* d_out, int out_size, void* d_ws, size_t ws_size,
                              hipStream_t stream) {
  const float* X  = (const float*)d_in[0];   // [B,128]
  const float* Nb = (const float*)d_in[1];   // [B,50,128]
  const float* W  = (const float*)d_in[2];   // [128,128]
  const float* Aw = (const float*)d_in[3];   // [256,128]
  const float* Ab = (const float*)d_in[4];   // [128]
  float* out = (float*)d_out;
  const int B = in_sizes[0] / DIM;

  // Workspace: 32KB W-frags | 32KB Wd-frags | 64KB Wc (f32) = 128KB total.
  unsigned short* wF  = (unsigned short*)d_ws;
  unsigned short* wdF = (unsigned short*)((char*)d_ws + 32768);
  float*          Wc  = (float*)((char*)d_ws + 65536);

  gat_prep<<<(DIM * DIM + 255) / 256, 256, 0, stream>>>(W, Aw, wF, wdF, Wc);
  gat_main<<<B, 256, 0, stream>>>(X, Nb, Ab, wF, wdF, Wc, out);
}
